// GAT_mod_75222057222986
// MI455X (gfx1250) — compile-verified, matrix operand order checked by reading, not run on hardware
//
#include <hip/hip_runtime.h>

// ---------------- problem constants (match reference) ----------------
constexpr int Nn = 50000;   // nodes
constexpr int Ee = 800000;  // edges
constexpr int Dd = 64;      // feature dim
constexpr int Hh = 4;       // heads
constexpr float NEG_SLOPE = 0.2f;
constexpr float BN_EPS = 1e-5f;

typedef __attribute__((ext_vector_type(16))) _Float16 v16h;
typedef __attribute__((ext_vector_type(8)))  _Float16 v8h;
typedef __attribute__((ext_vector_type(8)))  float    v8f;
typedef __attribute__((ext_vector_type(4)))  float    fx4;

__device__ __forceinline__ v8f wmma_f16(v16h a, v16h b, v8f c) {
  return __builtin_amdgcn_wmma_f32_16x16x32_f16(
      /*neg_a=*/false, a, /*neg_b=*/false, b,
      /*c_mod=*/(short)0, c, /*reuse_a=*/false, /*reuse_b=*/false);
}

// ---------------- A-tile loader (f32 -> f16, §7.12.2 layout) ----------
// lane<16 -> row=lane, K in {kc..kc+7, kc+16..kc+23};
// lane>=16 -> row=lane-16, K in {kc+8..kc+15, kc+24..kc+31}.
__device__ __forceinline__ v16h load_a_f32(const float* __restrict__ base,
                                           int pitch, int row0, int kc,
                                           int lane) {
  int r  = row0 + (lane & 15);
  int kh = (lane >> 4) & 1;
  const float* p = base + r * pitch + kc + kh * 8;
  fx4 x0 = ((const fx4*)p)[0];
  fx4 x1 = ((const fx4*)p)[1];
  fx4 y0 = ((const fx4*)(p + 16))[0];
  fx4 y1 = ((const fx4*)(p + 16))[1];
  v16h a;
#pragma unroll
  for (int j = 0; j < 4; ++j) {
    a[j]      = (_Float16)x0[j];
    a[4 + j]  = (_Float16)x1[j];
    a[8 + j]  = (_Float16)y0[j];
    a[12 + j] = (_Float16)y1[j];
  }
  return a;
}

// ---------------- weight prep: f32 -> pre-swizzled f16 B operands -----
// Table layout: tile-major [tile][lane][16 halves]; tile = col_tile*2+kcIdx.
// A consuming wave loads its operand as one contiguous 32B chunk per lane.
constexpr int WLIN_HALVES = 4 * 2 * 32 * 16;        // 4096  (64x64)
constexpr int WGAT_HALVES = 16 * 2 * 32 * 16;       // 16384 (256x64)

__global__ void prep_weights_kernel(const float* __restrict__ Wl,
                                    const float* __restrict__ Wg,
                                    _Float16* __restrict__ wlin_h,
                                    _Float16* __restrict__ wgat_h) {
  int i = blockIdx.x * blockDim.x + threadIdx.x;
  if (i >= WLIN_HALVES + WGAT_HALVES) return;
  const float* W;
  _Float16* out;
  int idx;
  if (i < WLIN_HALVES) { W = Wl; out = wlin_h; idx = i; }
  else                 { W = Wg; out = wgat_h; idx = i - WLIN_HALVES; }
  int t    = idx >> 9;                              // 32 lanes * 16 halves
  int lane = (idx >> 4) & 31;
  int j    = idx & 15;
  int col  = (t >> 1) * 16 + (lane & 15);           // output column = W row
  int kc   = (t & 1) * 32;
  int k    = kc + ((lane >> 4) & 1) * 8 + j + ((j >> 3) << 3);
  out[idx] = (_Float16)W[col * Dd + k];
}

// ---------------- fused GEMM1 + PReLU + GEMM2 + attention logits ------
// One wave per 16-row tile.  GEMM1 result is staged f16 in a wave-private
// LDS slab (DS ops from one wave are in-order -> no barrier), re-read in
// A-operand layout, then all 4 heads of GEMM2 run from registers.  The
// per-row attention dots are reduced from the accumulators with shfl_xor,
// so neither xp nor h is ever re-read from HBM.
// Loop unrolling is throttled (unroll<=2) so at most two 16-VGPR B
// operands are in flight -> no scratch spills.
__global__ void fused_gemm_kernel(const float* __restrict__ x,
                                  const _Float16* __restrict__ wlin_h,
                                  const _Float16* __restrict__ wgat_h,
                                  const float* __restrict__ bl,
                                  const float* __restrict__ pw,
                                  const float* __restrict__ att_s,
                                  const float* __restrict__ att_d,
                                  float* __restrict__ h,
                                  float* __restrict__ a_src,
                                  float* __restrict__ a_dst) {
  __shared__ _Float16 lds_xp[8][16 * Dd];           // 2KB per wave, 16KB/block
  int wave = (int)((blockIdx.x * blockDim.x + threadIdx.x) >> 5);
  int lane = threadIdx.x & 31;
  int wslot = threadIdx.x >> 5;
  if (wave >= Nn / 16) return;
  int row0 = wave * 16;
  int kh = lane >> 4, lc = lane & 15;
  _Float16* xl = lds_xp[wslot];

  // ---- GEMM1: xp = PReLU(x @ W_lin^T + b) -> LDS (f16) ----
  v16h a0 = load_a_f32(x, Dd, row0, 0, lane);
  v16h a1 = load_a_f32(x, Dd, row0, 32, lane);
  const v16h* blin = (const v16h*)wlin_h;
#pragma unroll 2
  for (int ct = 0; ct < 4; ++ct) {
    v16h b0 = blin[(ct * 2 + 0) * 32 + lane];
    v16h b1 = blin[(ct * 2 + 1) * 32 + lane];
    v8f acc = {};
    acc = wmma_f16(a0, b0, acc);
    acc = wmma_f16(a1, b1, acc);
    int col = ct * 16 + lc;
    float bb = bl[col], p = pw[col];
#pragma unroll
    for (int i = 0; i < 8; ++i) {
      float v = acc[i] + bb;
      v = (v >= 0.f) ? v : p * v;                   // per-channel PReLU
      xl[(i + 8 * kh) * Dd + col] = (_Float16)v;
    }
  }

  // ---- re-read xp from LDS in A-operand layout (wave-private) ----
  v16h c0, c1;
  {
    int r = lc;
    v8h p00 = *(const v8h*)(xl + r * Dd + 0  + kh * 8);
    v8h p01 = *(const v8h*)(xl + r * Dd + 16 + kh * 8);
    v8h p10 = *(const v8h*)(xl + r * Dd + 32 + kh * 8);
    v8h p11 = *(const v8h*)(xl + r * Dd + 48 + kh * 8);
#pragma unroll
    for (int j = 0; j < 8; ++j) {
      c0[j] = p00[j]; c0[8 + j] = p01[j];
      c1[j] = p10[j]; c1[8 + j] = p11[j];
    }
  }

  // ---- GEMM2 per head + fused attention dots ----
  const v16h* bgat = (const v16h*)wgat_h;
#pragma unroll 1
  for (int head = 0; head < Hh; ++head) {
    float sp[8], dp[8];
#pragma unroll
    for (int i = 0; i < 8; ++i) { sp[i] = 0.f; dp[i] = 0.f; }
#pragma unroll 2
    for (int ct = 0; ct < 4; ++ct) {
      int t = (head * 4 + ct) * 2;
      v16h b0 = bgat[(t + 0) * 32 + lane];
      v16h b1 = bgat[(t + 1) * 32 + lane];
      v8f acc = {};
      acc = wmma_f16(c0, b0, acc);
      acc = wmma_f16(c1, b1, acc);
      int col = head * Dd + ct * 16 + lc;           // 0..255
      float ws = att_s[col], wd = att_d[col];
#pragma unroll
      for (int i = 0; i < 8; ++i) {
        h[(size_t)(row0 + i + 8 * kh) * (Hh * Dd) + col] = acc[i];
        sp[i] += acc[i] * ws;
        dp[i] += acc[i] * wd;
      }
    }
    // reduce over the 16 lanes that share the same 8 rows
#pragma unroll
    for (int off = 1; off < 16; off <<= 1) {
#pragma unroll
      for (int i = 0; i < 8; ++i) {
        sp[i] += __shfl_xor(sp[i], off);
        dp[i] += __shfl_xor(dp[i], off);
      }
    }
    if (lc == 0) {                                  // lanes 0 and 16
#pragma unroll
      for (int i = 0; i < 8; ++i) {
        int r = row0 + i + 8 * kh;
        a_src[r * Hh + head] = sp[i];
        a_dst[r * Hh + head] = dp[i];
      }
    }
  }
}

// ---------------- order-preserving float<->uint for atomicMax ---------
__device__ __forceinline__ unsigned fenc(float f) {
  unsigned u = __float_as_uint(f);
  return (u >> 31) ? ~u : (u | 0x80000000u);
}
__device__ __forceinline__ float fdec(unsigned e) {
  return __uint_as_float((e >> 31) ? (e ^ 0x80000000u) : ~e);
}

__global__ void init_kernel(unsigned* __restrict__ emax,
                            float* __restrict__ denom,
                            float* __restrict__ accum,
                            float* __restrict__ stats) {
  int i = blockIdx.x * blockDim.x + threadIdx.x;
  if (i < Nn * Hh) { emax[i] = 0u; denom[i] = 0.f; }   // 0 < enc(any finite)
  if (i < Nn * Dd) accum[i] = 0.f;
  if (i < 128) stats[i] = 0.f;
}

__device__ __forceinline__ void edge_sd(const int* __restrict__ ei, int e,
                                        int& s, int& d) {
  if (e < Ee) { s = ei[e]; d = ei[Ee + e]; }
  else        { s = d = e - Ee; }                      // self loops
}

__global__ void edge_max_kernel(const int* __restrict__ ei,
                                const float* __restrict__ a_src,
                                const float* __restrict__ a_dst,
                                unsigned* __restrict__ emax) {
  int e = blockIdx.x * blockDim.x + threadIdx.x;
  if (e >= Ee + Nn) return;
  int s, d; edge_sd(ei, e, s, d);
#pragma unroll
  for (int hd = 0; hd < Hh; ++hd) {
    float v = a_src[s * Hh + hd] + a_dst[d * Hh + hd];
    v = (v >= 0.f) ? v : NEG_SLOPE * v;                // leaky relu
    atomicMax(&emax[d * Hh + hd], fenc(v));
  }
}

__global__ void edge_sum_kernel(const int* __restrict__ ei,
                                const float* __restrict__ a_src,
                                const float* __restrict__ a_dst,
                                const unsigned* __restrict__ emax,
                                float* __restrict__ denom) {
  int e = blockIdx.x * blockDim.x + threadIdx.x;
  if (e >= Ee + Nn) return;
  int s, d; edge_sd(ei, e, s, d);
#pragma unroll
  for (int hd = 0; hd < Hh; ++hd) {
    float v = a_src[s * Hh + hd] + a_dst[d * Hh + hd];
    v = (v >= 0.f) ? v : NEG_SLOPE * v;
    float p = __expf(v - fdec(emax[d * Hh + hd]));
    atomicAdd(&denom[d * Hh + hd], p);
  }
}

// One wave per edge; the 1/H head-mean is folded in so the scatter target
// is N x 64 (4x fewer atomics than N x H x 64).
__global__ void edge_agg_kernel(const int* __restrict__ ei,
                                const float* __restrict__ a_src,
                                const float* __restrict__ a_dst,
                                const unsigned* __restrict__ emax,
                                const float* __restrict__ denom,
                                const float* __restrict__ h,
                                float* __restrict__ accum) {
  int wave = (int)((blockIdx.x * blockDim.x + threadIdx.x) >> 5);
  int lane = threadIdx.x & 31;
  if (wave >= Ee + Nn) return;
  int s, d; edge_sd(ei, wave, s, d);
  const float* hrow = h + (size_t)s * (Hh * Dd);
  __builtin_prefetch(hrow + lane * 8, 0, 0);           // global_prefetch_b8
  float alpha[Hh];
#pragma unroll
  for (int hd = 0; hd < Hh; ++hd) {
    float v = a_src[s * Hh + hd] + a_dst[d * Hh + hd];
    v = (v >= 0.f) ? v : NEG_SLOPE * v;
    float p = __expf(v - fdec(emax[d * Hh + hd]));
    alpha[hd] = p / denom[d * Hh + hd] * (1.f / Hh);
  }
#pragma unroll
  for (int t = 0; t < 2; ++t) {
    int c = lane + t * 32;                             // feature channel
    float acc = 0.f;
#pragma unroll
    for (int hd = 0; hd < Hh; ++hd)
      acc += alpha[hd] * hrow[hd * Dd + c];
    atomicAdd(&accum[(size_t)d * Dd + c], acc);
  }
}

// ---------------- BatchNorm: stats (LDS reduce -> global atomics) -----
__global__ void bn_reduce_kernel(float* __restrict__ accum,
                                 const float* __restrict__ gbias,
                                 float* __restrict__ stats) {
  __shared__ float ssum[Dd];
  __shared__ float sssq[Dd];
  int tid = threadIdx.x;
  if (tid < Dd) { ssum[tid] = 0.f; sssq[tid] = 0.f; }
  __syncthreads();
  int i = blockIdx.x * blockDim.x + tid;
  if (i < Nn * Dd) {
    int c = i & (Dd - 1);
    float v = accum[i] + gbias[c];                     // head-mean + gat_bias
    accum[i] = v;                                      // keep for final pass
    atomicAdd(&ssum[c], v);
    atomicAdd(&sssq[c], v * v);
  }
  __syncthreads();
  if (tid < Dd) {
    atomicAdd(&stats[tid], ssum[tid]);
    atomicAdd(&stats[Dd + tid], sssq[tid]);
  }
}

__global__ void bn_final_kernel(const float* __restrict__ accum,
                                const float* __restrict__ stats,
                                const float* __restrict__ gamma,
                                const float* __restrict__ beta,
                                float* __restrict__ out) {
  int i = blockIdx.x * blockDim.x + threadIdx.x;
  if (i >= Nn * Dd) return;
  int c = i & (Dd - 1);
  float mean = stats[c] * (1.f / Nn);
  float var  = stats[Dd + c] * (1.f / Nn) - mean * mean;  // biased, as jnp.var
  float v = gamma[c] * (accum[i] - mean) * rsqrtf(var + BN_EPS) + beta[c];
  out[i] = (v > 0.f) ? v : 0.f;
}

// ---------------- host orchestration ---------------------------------
extern "C" void kernel_launch(void* const* d_in, const int* in_sizes, int n_in,
                              void* d_out, int out_size, void* d_ws, size_t ws_size,
                              hipStream_t stream) {
  const float* x     = (const float*)d_in[0];
  const int*   ei    = (const int*)d_in[1];
  const float* Wl    = (const float*)d_in[2];
  const float* bl    = (const float*)d_in[3];
  const float* pw    = (const float*)d_in[4];
  const float* Wg    = (const float*)d_in[5];
  const float* att_s = (const float*)d_in[6];
  const float* att_d = (const float*)d_in[7];
  const float* gbias = (const float*)d_in[8];
  const float* gam   = (const float*)d_in[9];
  const float* bet   = (const float*)d_in[10];
  float* out = (float*)d_out;

  _Float16* wlin_h = (_Float16*)d_ws;                      // 4096 halves
  _Float16* wgat_h = wlin_h + WLIN_HALVES;                 // 16384 halves
  float*    h      = (float*)(wgat_h + WGAT_HALVES);       // N*256
  float*    a_src  = h + (size_t)Nn * Hh * Dd;             // N*4
  float*    a_dst  = a_src + (size_t)Nn * Hh;              // N*4
  unsigned* emax   = (unsigned*)(a_dst + (size_t)Nn * Hh); // N*4
  float*    denom  = (float*)(emax + (size_t)Nn * Hh);     // N*4
  float*    accum  = denom + (size_t)Nn * Hh;              // N*64
  float*    stats  = accum + (size_t)Nn * Dd;              // 128

  const int BLK = 256;
  auto blocks = [](long long work, int per) { return (int)((work + per - 1) / per); };

  // init accumulators + pre-swizzle weights to f16 WMMA operand layout
  init_kernel<<<blocks((long long)Nn * Dd, BLK), BLK, 0, stream>>>(emax, denom, accum, stats);
  prep_weights_kernel<<<blocks(WLIN_HALVES + WGAT_HALVES, BLK), BLK, 0, stream>>>(
      Wl, Wg, wlin_h, wgat_h);
  // fused GEMM1 + PReLU + GEMM2 + attention logits (one wave / 16 rows)
  fused_gemm_kernel<<<blocks((long long)(Nn / 16) * 32, BLK), BLK, 0, stream>>>(
      x, wlin_h, wgat_h, bl, pw, att_s, att_d, h, a_src, a_dst);
  // segment softmax over destinations
  edge_max_kernel<<<blocks(Ee + Nn, BLK), BLK, 0, stream>>>(ei, a_src, a_dst, emax);
  edge_sum_kernel<<<blocks(Ee + Nn, BLK), BLK, 0, stream>>>(ei, a_src, a_dst, emax, denom);
  // aggregate messages (one wave / edge), head-mean folded in
  edge_agg_kernel<<<blocks((long long)(Ee + Nn) * 32, BLK), BLK, 0, stream>>>(
      ei, a_src, a_dst, emax, denom, h, accum);
  // batchnorm + relu
  bn_reduce_kernel<<<blocks((long long)Nn * Dd, BLK), BLK, 0, stream>>>(accum, gbias, stats);
  bn_final_kernel<<<blocks((long long)Nn * Dd, BLK), BLK, 0, stream>>>(accum, stats, gam, bet, out);
}
